// GatedDeltaNetVarlen_1477468750450
// MI455X (gfx1250) — matrix operand order repaired and checked by reading, not compiled, not run on hardware
//
#include <hip/hip_runtime.h>
#include <hip/hip_bf16.h>

// ---------------- problem constants (from reference) ----------------
#define HIDDEN   1024
#define HK       8
#define HV       16
#define DK       64
#define DV       64
#define KSZ      4
#define KEY_DIM  (HK * DK)                 // 512
#define VAL_DIM  (HV * DV)                 // 1024
#define CONV_DIM (2 * KEY_DIM + VAL_DIM)   // 2048
#define MZ_N     (CONV_DIM + VAL_DIM)      // 3072 (mixed || z fused GEMM width)
#define TOTAL    1536
#define NSEQ     4

typedef __attribute__((ext_vector_type(16))) __bf16 bf16x16;
typedef __attribute__((ext_vector_type(8)))  __bf16 bf16x8;
typedef __attribute__((ext_vector_type(8)))  float  f32x8;

// ---------------- helpers ----------------
__device__ __forceinline__ float sigmoidf_(float x) { return 1.f / (1.f + __expf(-x)); }
__device__ __forceinline__ float siluf_(float x)    { return x * sigmoidf_(x); }

// ---------------- conversion kernels ----------------
__global__ void k_cvt_bf16(const float* __restrict__ src, __bf16* __restrict__ dst, int n) {
    int i = blockIdx.x * blockDim.x + threadIdx.x;
    if (i < n) dst[i] = (__bf16)src[i];
}

// dst[k * dstStride + dstColOff + n] = (bf16) src[n * K + k]   (W row-major -> k-major)
__global__ void k_transpose_cvt(const float* __restrict__ src, __bf16* __restrict__ dst,
                                int Nsrc, int K, int dstStride, int dstColOff) {
    int i = blockIdx.x * blockDim.x + threadIdx.x;
    if (i >= Nsrc * K) return;
    int n = i / K, k = i - n * K;
    dst[(size_t)k * dstStride + dstColOff + n] = (__bf16)src[i];
}

// small b/a projections (TOTAL x 32 dots of length 1024) — latency-trivial, keep f32
__global__ void k_ba_proj(const float* __restrict__ hs, const float* __restrict__ Wb,
                          const float* __restrict__ Wa, float* __restrict__ ba) {
    int i = blockIdx.x * blockDim.x + threadIdx.x;
    if (i >= TOTAL * 32) return;
    int t = i >> 5, j = i & 31;
    const float* w = (j < HV) ? (Wb + (size_t)j * HIDDEN) : (Wa + (size_t)(j - HV) * HIDDEN);
    const float* x = hs + (size_t)t * HIDDEN;
    float acc = 0.f;
    for (int k = 0; k < HIDDEN; k += 4)
        acc += x[k]*w[k] + x[k+1]*w[k+1] + x[k+2]*w[k+2] + x[k+3]*w[k+3];
    ba[i] = acc;
}

// ---------------- WMMA bf16 GEMM: C(MxN,f32) = A(MxK,bf16 row-major) * Bt(KxN,bf16 k-major)
// 256 threads = 8 waves arranged 4(M) x 2(N); each wave computes a 16x64 strip with
// 4 f32x8 accumulators, reusing one A operand across 4 WMMAs (block tile 64 x 128).
// Operand fragments are double-buffered: iteration kb+32 is in flight while kb computes,
// so WMMA issue never waits on a fresh L2 round trip.
// A operand layout (ISA 7.12.2, 16-bit A 16x32): lane<16 -> M=lane, K={0..7,16..23};
// lane>=16 -> M=lane-16, K={8..15,24..31}. B operand: N = lane&15, 16 K values in the same order as A.
__global__ __launch_bounds__(256) void k_wmma_gemm(const __bf16* __restrict__ A,
                                                   const __bf16* __restrict__ Bt,
                                                   float* __restrict__ C,
                                                   int M, int N, int K) {
    const int tid  = threadIdx.x;
    const int wave = tid >> 5, lane = tid & 31;
    const int wm = wave & 3, wn = wave >> 2;
    const int m0 = blockIdx.y * 64 + wm * 16;
    const int n0 = blockIdx.x * 128 + wn * 64;
    const int row = lane & 15, hi = lane >> 4;

    f32x8 acc[4] = {};
    const __bf16* arow = A + (size_t)(m0 + row) * K + hi * 8;   // + kb -> K chunk base
    const __bf16* brow = Bt + (size_t)(hi * 8) * N + n0 + row;  // + kb*N -> k = kb + hi*8, n = n0 + row

    union Frag { bf16x16 v; bf16x8 h[2]; };
    Frag a[2], b[2][4];

    auto load_frags = [&](int buf, int kb) {
        a[buf].h[0] = *reinterpret_cast<const bf16x8*>(arow + kb);        // K = kb+hi*8..
        a[buf].h[1] = *reinterpret_cast<const bf16x8*>(arow + kb + 16);   // K = kb+16+hi*8..
        const __bf16* bp = brow + (size_t)kb * N;
        #pragma unroll
        for (int j = 0; j < 4; ++j) {
            #pragma unroll
            for (int s = 0; s < 8; ++s) {
                b[buf][j].v[s]     = bp[(size_t)s * N + j * 16];
                b[buf][j].v[s + 8] = bp[(size_t)(s + 16) * N + j * 16];
            }
        }
    };

    load_frags(0, 0);
    for (int kb = 0; kb < K; kb += 32) {
        const int cur = (kb >> 5) & 1;
        if (kb + 32 < K) load_frags(cur ^ 1, kb + 32);
        if (kb + 96 < K) {
            __builtin_prefetch(arow + kb + 96, 0, 1);
            __builtin_prefetch(brow + (size_t)(kb + 96) * N, 0, 1);
        }
        #pragma unroll
        for (int j = 0; j < 4; ++j)
            acc[j] = __builtin_amdgcn_wmma_f32_16x16x32_bf16(false, a[cur].v, false, b[cur][j].v,
                                                             (short)0, acc[j], false, false);
    }
    // C/D layout: VGPR r -> M = r + hi*8, N = lane&15
    #pragma unroll
    for (int j = 0; j < 4; ++j) {
        float* crow = C + (size_t)(m0 + hi * 8) * N + n0 + j * 16 + row;
        #pragma unroll
        for (int r = 0; r < 8; ++r) crow[(size_t)r * N] = acc[j][r];
    }
}

// ---------------- causal depthwise conv + SiLU + L2 norms + gates, one block per token
__global__ __launch_bounds__(256) void k_conv_norm(
        const float* __restrict__ MZ, const float* __restrict__ convw,
        const float* __restrict__ ba, const float* __restrict__ A_log,
        const float* __restrict__ dt_bias, const int* __restrict__ cu,
        float* __restrict__ qn, float* __restrict__ kn, float* __restrict__ vv,
        float* __restrict__ gg, float* __restrict__ bb) {
    __shared__ float smem[CONV_DIM];
    const int t = blockIdx.x;
    int s = 0;
    #pragma unroll
    for (int i = 1; i < NSEQ; ++i) if (t >= cu[i]) s = i;
    const int start = cu[s];

    for (int c = threadIdx.x; c < CONV_DIM; c += 256) {
        float acc = 0.f;
        #pragma unroll
        for (int j = 0; j < KSZ; ++j) {
            int tt = t - (KSZ - 1) + j;
            if (tt >= start) acc += convw[c * KSZ + j] * MZ[(size_t)tt * MZ_N + c];
        }
        smem[c] = siluf_(acc);
    }
    __syncthreads();

    // one wave per key head: L2-normalize q (and scale DK^-0.5) and k, expand GQA x2
    const int wave = threadIdx.x >> 5, lane = threadIdx.x & 31;
    {
        const int h = wave;                 // 0..7 key heads
        float q0 = smem[h * DK + lane],            q1 = smem[h * DK + lane + 32];
        float k0 = smem[KEY_DIM + h * DK + lane],  k1 = smem[KEY_DIM + h * DK + lane + 32];
        float sq = q0 * q0 + q1 * q1, sk = k0 * k0 + k1 * k1;
        #pragma unroll
        for (int m = 16; m >= 1; m >>= 1) { sq += __shfl_xor(sq, m, 32); sk += __shfl_xor(sk, m, 32); }
        float rq = rsqrtf(sq + 1e-6f) * 0.125f;   // * DK^-0.5
        float rk = rsqrtf(sk + 1e-6f);
        q0 *= rq; q1 *= rq; k0 *= rk; k1 *= rk;
        #pragma unroll
        for (int r = 0; r < 2; ++r) {
            int hv = h * 2 + r;
            float* qd = qn + ((size_t)t * HV + hv) * DK;
            float* kd = kn + ((size_t)t * HV + hv) * DK;
            qd[lane] = q0; qd[lane + 32] = q1;
            kd[lane] = k0; kd[lane + 32] = k1;
        }
    }
    for (int i = threadIdx.x; i < VAL_DIM; i += 256)
        vv[(size_t)t * VAL_DIM + i] = smem[2 * KEY_DIM + i];

    if (threadIdx.x < HV) {
        int h = threadIdx.x;
        float bv = ba[t * 32 + h];
        float av = ba[t * 32 + HV + h];
        bb[t * HV + h] = sigmoidf_(bv);
        float x  = av + dt_bias[h];
        float sp = (x > 20.f) ? x : log1pf(__expf(x));
        gg[t * HV + h] = -__expf(A_log[h]) * sp;
    }
}

// ---------------- gated delta rule scan: one block per (seq, head); lane owns one DV column.
// The per-step reductions are split into 4 independent FMA chains to shorten the serial
// dependence (this scan, not the GEMMs, bounds wall-clock time).
__global__ __launch_bounds__(64) void k_delta_rule(
        const float* __restrict__ qn, const float* __restrict__ kn,
        const float* __restrict__ vv, const float* __restrict__ gg,
        const float* __restrict__ bb, const int* __restrict__ cu,
        float* __restrict__ oo) {
    const int s = blockIdx.x >> 4;
    const int h = blockIdx.x & 15;
    const int n = threadIdx.x;          // DV column
    const int t0 = cu[s], t1 = cu[s + 1];
    __shared__ float qs[DK], ks[DK], gb[2];

    float S[DK];
    #pragma unroll
    for (int d = 0; d < DK; ++d) S[d] = 0.f;

    for (int t = t0; t < t1; ++t) {
        const size_t base = ((size_t)t * HV + h) * DK;
        qs[n] = qn[base + n];
        ks[n] = kn[base + n];
        if (n == 0) gb[0] = __expf(gg[t * HV + h]);
        if (n == 1) gb[1] = bb[t * HV + h];
        __syncthreads();
        const float eg = gb[0], beta = gb[1];

        float ks0 = 0.f, ks1 = 0.f, ks2 = 0.f, ks3 = 0.f;
        #pragma unroll
        for (int d = 0; d < DK; d += 4) {
            S[d]     *= eg; ks0 = fmaf(ks[d],     S[d],     ks0);
            S[d + 1] *= eg; ks1 = fmaf(ks[d + 1], S[d + 1], ks1);
            S[d + 2] *= eg; ks2 = fmaf(ks[d + 2], S[d + 2], ks2);
            S[d + 3] *= eg; ks3 = fmaf(ks[d + 3], S[d + 3], ks3);
        }
        const float ksum  = (ks0 + ks1) + (ks2 + ks3);
        const float delta = beta * (vv[base + n] - ksum);

        float o0 = 0.f, o1 = 0.f, o2 = 0.f, o3 = 0.f;
        #pragma unroll
        for (int d = 0; d < DK; d += 4) {
            S[d]     = fmaf(ks[d],     delta, S[d]);     o0 = fmaf(qs[d],     S[d],     o0);
            S[d + 1] = fmaf(ks[d + 1], delta, S[d + 1]); o1 = fmaf(qs[d + 1], S[d + 1], o1);
            S[d + 2] = fmaf(ks[d + 2], delta, S[d + 2]); o2 = fmaf(qs[d + 2], S[d + 2], o2);
            S[d + 3] = fmaf(ks[d + 3], delta, S[d + 3]); o3 = fmaf(qs[d + 3], S[d + 3], o3);
        }
        oo[base + n] = (o0 + o1) + (o2 + o3);
        __syncthreads();
    }
}

// ---------------- gated RMSNorm over DV + silu(z) gate; output bf16 for final GEMM
__global__ __launch_bounds__(256) void k_gate_norm(
        const float* __restrict__ oo, const float* __restrict__ MZ,
        const float* __restrict__ norm_w, __bf16* __restrict__ G) {
    const int t = blockIdx.x;
    const int h = threadIdx.x >> 4;
    const int i = threadIdx.x & 15;     // 4 dims per thread
    const size_t base = ((size_t)t * HV + h) * DV + i * 4;
    float ov[4];
    float ss = 0.f;
    #pragma unroll
    for (int d = 0; d < 4; ++d) { ov[d] = oo[base + d]; ss = fmaf(ov[d], ov[d], ss); }
    #pragma unroll
    for (int m = 8; m >= 1; m >>= 1) ss += __shfl_xor(ss, m, 16);
    const float r = rsqrtf(ss * (1.f / DV) + 1e-6f);
    const float* z = MZ + (size_t)t * MZ_N + CONV_DIM + h * DV + i * 4;
    __bf16* gp = G + (size_t)t * VAL_DIM + h * DV + i * 4;
    #pragma unroll
    for (int d = 0; d < 4; ++d)
        gp[d] = (__bf16)(ov[d] * r * norm_w[i * 4 + d] * siluf_(z[d]));
}

// ---------------- launcher ----------------
extern "C" void kernel_launch(void* const* d_in, const int* in_sizes, int n_in,
                              void* d_out, int out_size, void* d_ws, size_t ws_size,
                              hipStream_t stream) {
    const float* hidden = (const float*)d_in[0];
    const float* W_qkv  = (const float*)d_in[1];
    const float* W_z    = (const float*)d_in[2];
    const float* W_b    = (const float*)d_in[3];
    const float* W_a    = (const float*)d_in[4];
    const float* convw  = (const float*)d_in[5];
    const float* dtb    = (const float*)d_in[6];
    const float* A_log  = (const float*)d_in[7];
    const float* nw     = (const float*)d_in[8];
    const float* W_out  = (const float*)d_in[9];
    const int*   cu     = (const int*)d_in[10];
    (void)in_sizes; (void)n_in; (void)out_size; (void)ws_size;

    // workspace carve-up (256B aligned chunks)
    char* w = (char*)d_ws;
    auto take = [&](size_t bytes) { char* p = w; w += (bytes + 255) & ~(size_t)255; return p; };
    __bf16* hbf = (__bf16*)take((size_t)TOTAL * HIDDEN * 2);
    __bf16* WtA = (__bf16*)take((size_t)HIDDEN * MZ_N * 2);     // [W_qkv || W_z] k-major
    __bf16* WtO = (__bf16*)take((size_t)HIDDEN * VAL_DIM * 2);  // W_out k-major
    float*  MZ  = (float*) take((size_t)TOTAL * MZ_N * 4);
    float*  ba  = (float*) take((size_t)TOTAL * 32 * 4);
    float*  qn  = (float*) take((size_t)TOTAL * HV * DK * 4);
    float*  kn  = (float*) take((size_t)TOTAL * HV * DK * 4);
    float*  vv  = (float*) take((size_t)TOTAL * HV * DV * 4);
    float*  gg  = (float*) take((size_t)TOTAL * HV * 4);
    float*  bb  = (float*) take((size_t)TOTAL * HV * 4);
    float*  oo  = (float*) take((size_t)TOTAL * HV * DV * 4);
    __bf16* G   = (__bf16*)take((size_t)TOTAL * VAL_DIM * 2);

    // 1) precision conversion + weight transposes (k-major for contiguous WMMA B loads)
    k_cvt_bf16<<<(TOTAL * HIDDEN + 255) / 256, 256, 0, stream>>>(hidden, hbf, TOTAL * HIDDEN);
    k_transpose_cvt<<<(CONV_DIM * HIDDEN + 255) / 256, 256, 0, stream>>>(W_qkv, WtA, CONV_DIM, HIDDEN, MZ_N, 0);
    k_transpose_cvt<<<(VAL_DIM * HIDDEN + 255) / 256, 256, 0, stream>>>(W_z, WtA, VAL_DIM, HIDDEN, MZ_N, CONV_DIM);
    k_transpose_cvt<<<(HIDDEN * VAL_DIM + 255) / 256, 256, 0, stream>>>(W_out, WtO, HIDDEN, VAL_DIM, VAL_DIM, 0);
    k_ba_proj<<<(TOTAL * 32 + 255) / 256, 256, 0, stream>>>(hidden, W_b, W_a, ba);

    // 2) fused projection GEMM: MZ = hidden * [W_qkv||W_z]^T   (1536 x 3072 x 1024)
    dim3 g1(MZ_N / 128, TOTAL / 64);
    k_wmma_gemm<<<g1, 256, 0, stream>>>(hbf, WtA, MZ, TOTAL, MZ_N, HIDDEN);

    // 3) conv + silu + L2 norms + gates
    k_conv_norm<<<TOTAL, 256, 0, stream>>>(MZ, convw, ba, A_log, dtb, cu, qn, kn, vv, gg, bb);

    // 4) sequential gated delta rule (64 independent chains)
    k_delta_rule<<<NSEQ * HV, 64, 0, stream>>>(qn, kn, vv, gg, bb, cu, oo);

    // 5) gated RMSNorm -> bf16
    k_gate_norm<<<TOTAL, 256, 0, stream>>>(oo, MZ, nw, G);

    // 6) output GEMM: d_out = G * W_out^T   (1536 x 1024 x 1024)
    dim3 g2(VAL_DIM / 128, TOTAL / 64);
    k_wmma_gemm<<<g2, 256, 0, stream>>>(G, WtO, (float*)d_out, TOTAL, VAL_DIM, HIDDEN);
}
